// DeepAR_29764123361432
// MI455X (gfx1250) — matrix operand order repaired and checked by reading, compiled, not run on hardware
//
#include <hip/hip_runtime.h>

// Problem constants (match reference)
#define B_   128
#define T_   160
#define F_   512
#define E_   64
#define H_   1024
#define IN0_ 576          // F_ + E_
#define G4_  4096         // 4*H_

typedef __attribute__((ext_vector_type(16))) __bf16 bf16x16;
typedef __attribute__((ext_vector_type(8)))  float  v8f;

// ---------- helpers ----------
__device__ __forceinline__ unsigned short f2bf(float x) {
    unsigned u = __float_as_uint(x);
    unsigned r = 0x7fffu + ((u >> 16) & 1u);   // round-to-nearest-even
    return (unsigned short)((u + r) >> 16);
}
__device__ __forceinline__ float sigm(float x) { return 1.0f / (1.0f + __expf(-x)); }

// ---------- one-time prep kernels ----------
__global__ void cvt_bf16_kernel(const float* __restrict__ s, unsigned short* __restrict__ d, int n) {
    int i = blockIdx.x * 256 + threadIdx.x;
    if (i < n) d[i] = f2bf(s[i]);
}
__global__ void add_bias_kernel(const float* __restrict__ a, const float* __restrict__ b,
                                float* __restrict__ d, int n) {
    int i = blockIdx.x * 256 + threadIdx.x;
    if (i < n) d[i] = a[i] + b[i];
}
__global__ void zero_f32_kernel(float* __restrict__ p, int n) {
    int i = blockIdx.x * 256 + threadIdx.x;
    if (i < n) p[i] = 0.0f;
}
__global__ void zero_u16_kernel(unsigned short* __restrict__ p, int n) {
    int i = blockIdx.x * 256 + threadIdx.x;
    if (i < n) p[i] = 0;
}

// ---------- per-step: build x = [feat_t , label-or-mu-embed] as bf16 ----------
__global__ void assemble_x_kernel(const float* __restrict__ feat, const float* __restrict__ emb,
                                  const unsigned char* __restrict__ mask,
                                  const float* __restrict__ W_tgt, const float* __restrict__ b_tgt,
                                  const float* __restrict__ mu_state,
                                  unsigned short* __restrict__ xbf, int t) {
    int idx = blockIdx.x * 256 + threadIdx.x;
    if (idx >= B_ * IN0_) return;
    int b = idx / IN0_;
    int j = idx - b * IN0_;
    float v;
    if (j < F_) {
        v = feat[((size_t)b * T_ + t) * F_ + j];
    } else {
        int e = j - F_;
        bool use_mu = (t > 0) && (mask[b * T_ + t] != 0);
        v = use_mu ? fmaf(mu_state[b], W_tgt[e], b_tgt[e])
                   : emb[((size_t)b * T_ + t) * E_ + e];
    }
    xbf[idx] = f2bf(v);
}

// ---------- WMMA GEMM: G[128,4096] = A0[128,K0]*W0^T + A1[128,K1]*W1^T + bias ----------
// A (16-bit 16x32) lane layout: lane<16 -> row M=lane, K {0..15};       lane>=16 -> K {16..31}
// B = W^T (32x16):             lane<16 -> col N=lane, K {0..15};       lane>=16 -> K {16..31}
__device__ __forceinline__ void gemm_seg(v8f& acc,
                                         const unsigned short* __restrict__ Arow,
                                         const unsigned short* __restrict__ Wrow,
                                         int K, int asel) {
    union Frag { bf16x16 v; float4 q[2]; };
#pragma unroll 4
    for (int k0 = 0; k0 < K; k0 += 32) {
        Frag a, b;
        a.q[0] = *reinterpret_cast<const float4*>(Arow + k0 + asel * 16);
        a.q[1] = *reinterpret_cast<const float4*>(Arow + k0 + asel * 16 + 8);
        b.q[0] = *reinterpret_cast<const float4*>(Wrow + k0);
        b.q[1] = *reinterpret_cast<const float4*>(Wrow + k0 + 8);
        acc = __builtin_amdgcn_wmma_f32_16x16x32_bf16(false, a.v, false, b.v,
                                                      (short)0, acc, false, false);
    }
}

__global__ __launch_bounds__(256) void gates_gemm_kernel(
        const unsigned short* __restrict__ A0, int K0, const unsigned short* __restrict__ W0,
        const unsigned short* __restrict__ A1, int K1, const unsigned short* __restrict__ W1,
        const float* __restrict__ bias, float* __restrict__ G) {
    const int lane   = threadIdx.x & 31;
    const int wave   = threadIdx.x >> 5;
    const int tile   = blockIdx.x * 8 + wave;   // 0..2047
    const int nt     = tile & 255;              // N tile (4096/16)
    const int mt     = tile >> 8;               // M tile (128/16) — constant per block
    const int n0     = nt << 4;
    const int m0     = mt << 4;
    const int lane16 = lane & 15;
    const int hsel   = lane >> 4;               // half-wave selector

    const float bb = bias[n0 + lane16];
    v8f acc;
#pragma unroll
    for (int r = 0; r < 8; ++r) acc[r] = bb;

    const int arow = m0 + lane16;               // A row (M index)
    const int brow = n0 + lane16;               // W row (N index)

    gemm_seg(acc, A0 + (size_t)arow * K0, W0 + (size_t)brow * K0 + hsel * 16, K0, hsel);
    gemm_seg(acc, A1 + (size_t)arow * K1, W1 + (size_t)brow * K1 + hsel * 16, K1, hsel);

    float* Gp = G + (size_t)(m0 + hsel * 8) * G4_ + n0 + lane16;
#pragma unroll
    for (int r = 0; r < 8; ++r) Gp[(size_t)r * G4_] = acc[r];
}

// ---------- LSTM cell pointwise update ----------
__global__ void lstm_cell_kernel(const float* __restrict__ G, float* __restrict__ h,
                                 float* __restrict__ c, unsigned short* __restrict__ hbf) {
    int idx = blockIdx.x * 256 + threadIdx.x;
    if (idx >= B_ * H_) return;
    int b = idx >> 10;                 // / H_
    int k = idx & (H_ - 1);
    const float* g = G + (size_t)b * G4_;
    float gi = g[k], gf = g[H_ + k], gg = g[2 * H_ + k], go = g[3 * H_ + k];
    float cc = sigm(gf) * c[idx] + sigm(gi) * tanhf(gg);
    float hh = sigm(go) * tanhf(cc);
    c[idx]   = cc;
    h[idx]   = hh;
    hbf[idx] = f2bf(hh);
}

// ---------- mu / sigma head: hp[b,2k]=h0[b,k], hp[b,2k+1]=h1[b,k] ----------
__global__ __launch_bounds__(256) void head_kernel(
        const float* __restrict__ h0, const float* __restrict__ h1,
        const float* __restrict__ mu_w, const float* __restrict__ mu_b,
        const float* __restrict__ sig_w, const float* __restrict__ sig_b,
        float* __restrict__ mu_state, float* __restrict__ out, int t) {
    __shared__ float s0[256], s1[256];
    int b = blockIdx.x;
    float a0 = 0.0f, a1 = 0.0f;
    for (int k = threadIdx.x; k < H_; k += 256) {
        float v0 = h0[(size_t)b * H_ + k];
        float v1 = h1[(size_t)b * H_ + k];
        a0 = fmaf(v0, mu_w[2 * k],  fmaf(v1, mu_w[2 * k + 1],  a0));
        a1 = fmaf(v0, sig_w[2 * k], fmaf(v1, sig_w[2 * k + 1], a1));
    }
    s0[threadIdx.x] = a0;
    s1[threadIdx.x] = a1;
    __syncthreads();
    for (int st = 128; st > 0; st >>= 1) {
        if (threadIdx.x < st) {
            s0[threadIdx.x] += s0[threadIdx.x + st];
            s1[threadIdx.x] += s1[threadIdx.x + st];
        }
        __syncthreads();
    }
    if (threadIdx.x == 0) {
        float mu = s0[0] + mu_b[0];
        float sp = s1[0] + sig_b[0];
        float sg = fmaxf(sp, 0.0f) + log1pf(__expf(-fabsf(sp)));  // stable softplus
        mu_state[b] = mu;
        out[((size_t)b * T_ + t) * 2 + 0] = mu;
        out[((size_t)b * T_ + t) * 2 + 1] = sg;
    }
}

// ---------- host driver ----------
extern "C" void kernel_launch(void* const* d_in, const int* in_sizes, int n_in,
                              void* d_out, int out_size, void* d_ws, size_t ws_size,
                              hipStream_t stream) {
    const float* feat  = (const float*)d_in[0];
    const float* emb   = (const float*)d_in[1];
    const unsigned char* mask = (const unsigned char*)d_in[2];
    const float* W_ih0 = (const float*)d_in[3];
    const float* W_hh0 = (const float*)d_in[4];
    const float* b_ih0 = (const float*)d_in[5];
    const float* b_hh0 = (const float*)d_in[6];
    const float* W_ih1 = (const float*)d_in[7];
    const float* W_hh1 = (const float*)d_in[8];
    const float* b_ih1 = (const float*)d_in[9];
    const float* b_hh1 = (const float*)d_in[10];
    const float* W_tgt = (const float*)d_in[11];
    const float* b_tgt = (const float*)d_in[12];
    const float* mu_w  = (const float*)d_in[13];
    const float* mu_b  = (const float*)d_in[14];
    const float* sig_w = (const float*)d_in[15];
    const float* sig_b = (const float*)d_in[16];
    float* out = (float*)d_out;

    // workspace carve-up (256-B aligned slices)
    char* ws = (char*)d_ws;
    size_t off = 0;
    auto take = [&](size_t bytes) {
        char* p = ws + off;
        off += (bytes + 255) & ~(size_t)255;
        return (void*)p;
    };
    unsigned short* Wih0b = (unsigned short*)take((size_t)G4_ * IN0_ * 2);
    unsigned short* Whh0b = (unsigned short*)take((size_t)G4_ * H_ * 2);
    unsigned short* Wih1b = (unsigned short*)take((size_t)G4_ * H_ * 2);
    unsigned short* Whh1b = (unsigned short*)take((size_t)G4_ * H_ * 2);
    float* bsum0 = (float*)take((size_t)G4_ * 4);
    float* bsum1 = (float*)take((size_t)G4_ * 4);
    float* states = (float*)take(((size_t)4 * B_ * H_ + B_) * 4);  // h0,c0,h1,c1,mu
    float* h0 = states;
    float* c0 = h0 + B_ * H_;
    float* h1 = c0 + B_ * H_;
    float* c1 = h1 + B_ * H_;
    float* mu_state = c1 + B_ * H_;
    unsigned short* hbf = (unsigned short*)take((size_t)2 * B_ * H_ * 2);
    unsigned short* h0b = hbf;
    unsigned short* h1b = hbf + B_ * H_;
    unsigned short* xbf = (unsigned short*)take((size_t)B_ * IN0_ * 2);
    float* gates = (float*)take((size_t)B_ * G4_ * 4);

    const int TPB = 256;
    auto cdiv = [](int a, int b) { return (a + b - 1) / b; };

    // one-time per-call prep: bf16 weights, bias sums, zeroed state
    cvt_bf16_kernel<<<cdiv(G4_ * IN0_, TPB), TPB, 0, stream>>>(W_ih0, Wih0b, G4_ * IN0_);
    cvt_bf16_kernel<<<cdiv(G4_ * H_,  TPB), TPB, 0, stream>>>(W_hh0, Whh0b, G4_ * H_);
    cvt_bf16_kernel<<<cdiv(G4_ * H_,  TPB), TPB, 0, stream>>>(W_ih1, Wih1b, G4_ * H_);
    cvt_bf16_kernel<<<cdiv(G4_ * H_,  TPB), TPB, 0, stream>>>(W_hh1, Whh1b, G4_ * H_);
    add_bias_kernel<<<cdiv(G4_, TPB), TPB, 0, stream>>>(b_ih0, b_hh0, bsum0, G4_);
    add_bias_kernel<<<cdiv(G4_, TPB), TPB, 0, stream>>>(b_ih1, b_hh1, bsum1, G4_);
    zero_f32_kernel<<<cdiv(4 * B_ * H_ + B_, TPB), TPB, 0, stream>>>(states, 4 * B_ * H_ + B_);
    zero_u16_kernel<<<cdiv(2 * B_ * H_, TPB), TPB, 0, stream>>>(hbf, 2 * B_ * H_);

    // sequential decode loop
    for (int t = 0; t < T_; ++t) {
        assemble_x_kernel<<<cdiv(B_ * IN0_, TPB), TPB, 0, stream>>>(
            feat, emb, mask, W_tgt, b_tgt, mu_state, xbf, t);
        gates_gemm_kernel<<<256, 256, 0, stream>>>(
            xbf, IN0_, Wih0b, h0b, H_, Whh0b, bsum0, gates);
        lstm_cell_kernel<<<cdiv(B_ * H_, TPB), TPB, 0, stream>>>(gates, h0, c0, h0b);
        gates_gemm_kernel<<<256, 256, 0, stream>>>(
            h0b, H_, Wih1b, h1b, H_, Whh1b, bsum1, gates);
        lstm_cell_kernel<<<cdiv(B_ * H_, TPB), TPB, 0, stream>>>(gates, h1, c1, h1b);
        head_kernel<<<B_, 256, 0, stream>>>(
            h0, h1, mu_w, mu_b, sig_w, sig_b, mu_state, out, t);
    }
}
